// RandomizedOscillatorsNetwork_7035156431233
// MI455X (gfx1250) — compile-verified, matrix operand order checked by reading, not run on hardware
//
#include <hip/hip_runtime.h>
#include <hip/hip_bf16.h>
#include <math.h>

typedef __attribute__((ext_vector_type(16))) __bf16 v16bf;
typedef __attribute__((ext_vector_type(8)))  __bf16 v8bf;
typedef __attribute__((ext_vector_type(8)))  float  v8f;
typedef __attribute__((ext_vector_type(4)))  unsigned int u32x4;
typedef __attribute__((ext_vector_type(4)))  int i32x4;
typedef __attribute__((ext_vector_type(8)))  int i32x8;

#define RON_B 128
#define RON_T 1024
#define RON_I 64
#define RON_H 512
#define RON_DT 0.042f

#ifndef __has_builtin
#define __has_builtin(x) 0
#endif
#if __has_builtin(__builtin_amdgcn_tensor_load_to_lds) && \
    __has_builtin(__builtin_amdgcn_s_wait_tensorcnt)
#define RON_HAVE_TDM 1
#else
#define RON_HAVE_TDM 0
#endif

// ---------------------------------------------------------------------------
// Prep: convert + transpose weights to bf16 column-major so WMMA B-matrix
// fragments are K-contiguous per lane.
//   h2hT[n*512 + k] = (bf16) h2h[k*512 + n]   (512x512)
//   x2hT[n*64  + k] = (bf16) x2h[k*512 + n]   (512x64)
// ---------------------------------------------------------------------------
__global__ __launch_bounds__(256) void ron_prep_kernel(
    const float* __restrict__ x2h, const float* __restrict__ h2h,
    __bf16* __restrict__ x2hT, __bf16* __restrict__ h2hT) {
  int idx = blockIdx.x * 256 + threadIdx.x;
  if (idx < RON_H * RON_I) {
    int n = idx >> 6, k = idx & 63;
    x2hT[idx] = (__bf16)x2h[k * RON_H + n];
  }
  if (idx < RON_H * RON_H) {
    int n = idx >> 9, k = idx & 511;
    h2hT[idx] = (__bf16)h2h[k * RON_H + n];
  }
}

// 16x32 bf16 fragment for v_wmma_f32_16x16x32_bf16, from a K-contiguous row.
// elements 0..7 <- K=kbase..+7 ; elements 8..15 <- K=kbase+16..+23
__device__ __forceinline__ v16bf ron_load_frag(const __bf16* base_row, int kbase) {
  union { v16bf v; struct { v8bf lo; v8bf hi; } s; } u;
  u.s.lo = *(const v8bf*)(base_row + kbase);
  u.s.hi = *(const v8bf*)(base_row + kbase + 16);
  return u.v;
}

// Same fragment but converting from an fp32 row (x tile staged raw by TDM).
__device__ __forceinline__ v16bf ron_load_frag_f32(const float* base_row, int kbase) {
  v16bf r;
#pragma unroll
  for (int i = 0; i < 8; ++i) r[i] = (__bf16)base_row[kbase + i];
#pragma unroll
  for (int i = 0; i < 8; ++i) r[i + 8] = (__bf16)base_row[kbase + 16 + i];
  return r;
}

// Issue a TDM 2D tile load: 16 rows x 64 fp32, row stride T*I elements,
// from global x into LDS. D# per CDNA5 ISA §8.3/8.4; groups 2/3 unused (2D).
__device__ __forceinline__ void ron_tdm_load_x(const float* gsrc, void* lds_dst) {
#if RON_HAVE_TDM
  unsigned long long ga = (unsigned long long)(uintptr_t)gsrc;
  unsigned int lds_off = (unsigned int)(uintptr_t)lds_dst;  // addr[31:0] = LDS byte offset
  u32x4 g0;
  g0[0] = 1u;                                   // count=1 valid user D#
  g0[1] = lds_off;                              // lds_addr
  g0[2] = (unsigned int)ga;                     // global_addr[31:0]
  g0[3] = (unsigned int)((ga >> 32) & 0x1FFFFFFu) | 0x80000000u;  // [56:32] | type=2
  i32x8 g1;
  g1[0] = 0x00020000;                           // data_size=2 (4 bytes)
  g1[1] = RON_I << 16;                          // tensor_dim0 = 64
  g1[2] = 16 << 16;                             // tensor_dim1 = 16
  g1[3] = RON_I << 16;                          // tile_dim0 = 64
  g1[4] = 16;                                   // tile_dim1 = 16, tile_dim2 = 0
  g1[5] = RON_T * RON_I;                        // tensor_dim0_stride = 65536
  g1[6] = 0;
  g1[7] = 0;
  i32x4 z4 = {0, 0, 0, 0};
#if __clang_major__ >= 23
  i32x8 z8 = {0, 0, 0, 0, 0, 0, 0, 0};
  __builtin_amdgcn_tensor_load_to_lds(g0, g1, z4, z4, z8, 0);
#else
  __builtin_amdgcn_tensor_load_to_lds(g0, g1, z4, z4, 0);
#endif
#else
  (void)gsrc; (void)lds_dst;
#endif
}

// ---------------------------------------------------------------------------
// Persistent recurrence kernel.
// grid = B/16 = 8 workgroups, block = 512 threads (16 waves), one WGP each.
// Wave w owns output columns [w*32, w*32+32) as two 16x16 WMMA tiles.
// ---------------------------------------------------------------------------
__global__ __launch_bounds__(512) void ron_scan_kernel(
    const float* __restrict__ x,        // [B,T,I] fp32
    const __bf16* __restrict__ h2hT,    // [H][H] col-major bf16 (L2-resident)
    const __bf16* __restrict__ x2hT,    // [H][I] col-major bf16 (cache-hot)
    const float* __restrict__ bias,
    const float* __restrict__ gam,
    const float* __restrict__ eps,
    float* __restrict__ out) {          // states [B,T,H] ++ last hy [B,H]
  __shared__ __bf16 shy[2][16 * RON_H];   // hy tile bf16, double buffered (32KB)
  __shared__ float  sx[2][16 * RON_I];    // x_t tile fp32, double buffered (8KB)

  const int tid  = threadIdx.x;
  const int lane = tid & 31;
  const int wave = tid >> 5;
  const int b0   = blockIdx.x * 16;
  const int lcol = lane & 15;             // A: row M%16 ; B: col N%16 ; C: col N
  const int half = lane >> 4;
  const int kb_off = half << 3;           // per-half K chunk base: 0 or 8
  const int Mbase  = half << 3;           // C/D rows: M = Mbase + r
  const int n0 = wave * 32;

  // --- one-time setup -------------------------------------------------------
#if RON_HAVE_TDM
  if (wave == 0)
    ron_tdm_load_x(x + (size_t)b0 * RON_T * RON_I, (void*)&sx[0][0]);
#else
  {
    int e = tid;
#pragma unroll
    for (int rep = 0; rep < 2; ++rep, e += 512) {
      int row = e >> 6, k = e & 63;
      sx[0][e] = x[(((size_t)(b0 + row)) * RON_T + 0) * RON_I + k];
    }
  }
#endif
  for (int i = tid; i < 16 * RON_H; i += 512) shy[0][i] = (__bf16)0.0f;

  float pb[2], pg[2], pe[2];
#pragma unroll
  for (int tt = 0; tt < 2; ++tt) {
    int n = n0 + tt * 16 + lcol;
    pb[tt] = bias[n];
    pg[tt] = gam[n];
    pe[tt] = eps[n];
  }

  float hy[2][8], hz[2][8];
#pragma unroll
  for (int tt = 0; tt < 2; ++tt)
#pragma unroll
    for (int r = 0; r < 8; ++r) { hy[tt][r] = 0.0f; hz[tt][r] = 0.0f; }

  float* outLast = out + (size_t)RON_B * RON_T * RON_H;

#if RON_HAVE_TDM
  if (wave == 0) __builtin_amdgcn_s_wait_tensorcnt(0);
#endif
  __syncthreads();

  // --- time scan ------------------------------------------------------------
  int cur = 0;
  for (int t = 0; t < RON_T; ++t) {
    const int nxt = cur ^ 1;

#if RON_HAVE_TDM
    // Kick off next step's x tile DMA; it overlaps this entire step.
    if (wave == 0 && t + 1 < RON_T)
      ron_tdm_load_x(x + ((size_t)b0 * RON_T + (t + 1)) * RON_I,
                     (void*)&sx[nxt][0]);
#endif

    // 4 independent accumulation chains: [tile][even/odd-k] for XDL pipelining
    v8f acc[2][2];
    acc[0][0] = (v8f){0.f, 0.f, 0.f, 0.f, 0.f, 0.f, 0.f, 0.f};
    acc[0][1] = acc[0][0]; acc[1][0] = acc[0][0]; acc[1][1] = acc[0][0];

    // (1) u_t contribution: A = x_t [16x64] (fp32 LDS, cvt), B = x2hT, K=64
#pragma unroll
    for (int kk = 0; kk < 2; ++kk) {
      v16bf a = ron_load_frag_f32(&sx[cur][lcol * RON_I], kk * 32 + kb_off);
#pragma unroll
      for (int tt = 0; tt < 2; ++tt) {
        v16bf b = ron_load_frag(x2hT + (size_t)(n0 + tt * 16 + lcol) * RON_I,
                                kk * 32 + kb_off);
        acc[tt][kk] = __builtin_amdgcn_wmma_f32_16x16x32_bf16(
            false, a, false, b, (short)0, acc[tt][kk], false, false);
      }
    }

    // (2) recurrent GEMM: A = hy [16x512] (LDS bf16), B = h2hT (L2), K=512
#pragma unroll 4
    for (int kk = 0; kk < 16; ++kk) {
      v16bf a = ron_load_frag(&shy[cur][lcol * RON_H], kk * 32 + kb_off);
#pragma unroll
      for (int tt = 0; tt < 2; ++tt) {
        v16bf b = ron_load_frag(h2hT + (size_t)(n0 + tt * 16 + lcol) * RON_H,
                                kk * 32 + kb_off);
        acc[tt][kk & 1] = __builtin_amdgcn_wmma_f32_16x16x32_bf16(
            false, a, false, b, (short)0, acc[tt][kk & 1], false, false);
      }
    }

    // (3) elementwise oscillator update; state lives in C-matrix layout
#pragma unroll
    for (int tt = 0; tt < 2; ++tt) {
      const int n = n0 + tt * 16 + lcol;
      v8f s = acc[tt][0] + acc[tt][1];
#pragma unroll
      for (int r = 0; r < 8; ++r) {
        const int M = Mbase + r;
        float w = tanhf(s[r] + pb[tt]);
        float z = hz[tt][r];
        float y = hy[tt][r];
        z = z + RON_DT * (w - pg[tt] * y - pe[tt] * z);
        y = y + RON_DT * z;
        hz[tt][r] = z;
        hy[tt][r] = y;
        out[(((size_t)(b0 + M)) * RON_T + t) * RON_H + n] = y;
        shy[nxt][M * RON_H + n] = (__bf16)y;
        if (t == RON_T - 1) outLast[(size_t)(b0 + M) * RON_H + n] = y;
      }
    }

#if !RON_HAVE_TDM
    if (t + 1 < RON_T) {
      int e = tid;
#pragma unroll
      for (int rep = 0; rep < 2; ++rep, e += 512) {
        int row = e >> 6, k = e & 63;
        sx[nxt][e] = x[(((size_t)(b0 + row)) * RON_T + (t + 1)) * RON_I + k];
      }
    }
#else
    if (wave == 0) __builtin_amdgcn_s_wait_tensorcnt(0);
#endif
    __syncthreads();
    cur = nxt;
  }
}

// ---------------------------------------------------------------------------
extern "C" void kernel_launch(void* const* d_in, const int* in_sizes, int n_in,
                              void* d_out, int out_size, void* d_ws, size_t ws_size,
                              hipStream_t stream) {
  (void)in_sizes; (void)n_in; (void)out_size; (void)ws_size;
  const float* x    = (const float*)d_in[0];
  const float* x2h  = (const float*)d_in[1];
  const float* h2h  = (const float*)d_in[2];
  const float* bias = (const float*)d_in[3];
  const float* gam  = (const float*)d_in[4];
  const float* eps  = (const float*)d_in[5];
  float* out = (float*)d_out;

  __bf16* h2hT = (__bf16*)d_ws;                  // 512*512 bf16 = 512 KB
  __bf16* x2hT = h2hT + RON_H * RON_H;           // 512*64  bf16 =  64 KB

  ron_prep_kernel<<<(RON_H * RON_H) / 256, 256, 0, stream>>>(x2h, h2h, x2hT, h2hT);
  ron_scan_kernel<<<RON_B / 16, 512, 0, stream>>>(x, h2hT, x2hT, bias, gam, eps, out);
}
